// SpatialMemory_38156489458113
// MI455X (gfx1250) — matrix operand order repaired and checked by reading, not compiled, not run on hardware
//
#include <hip/hip_runtime.h>

#define M_SIZE 8192
#define F_DIM  512
#define N_Q    32768

typedef __attribute__((ext_vector_type(16))) __bf16 v16bf;
typedef __attribute__((ext_vector_type(8)))  float  v8f;
typedef unsigned short u16;
typedef unsigned int   u32;

union V16 {
  v16bf v;
  u16   h[16];
  uint4 q[2];
};

// float -> bf16 with round-to-nearest-even, pure bit manipulation.
__device__ inline u16 f2bf(float f) {
  u32 u = __float_as_uint(f);
  u32 r = u + 0x7FFFu + ((u >> 16) & 1u);
  return (u16)(r >> 16);
}

// ---------------------------------------------------------------------------
// Kernel 1: normalize positions & position_embeddings, convert features->bf16
// ---------------------------------------------------------------------------
__global__ __launch_bounds__(256)
void prep_kernel(const float* __restrict__ feat,
                 const float* __restrict__ pos,
                 const float* __restrict__ pemb,
                 u16*  __restrict__ featbf,
                 float* __restrict__ posn,
                 float* __restrict__ embn) {
  int i = blockIdx.x * blockDim.x + threadIdx.x;
  if (i < M_SIZE * F_DIM) featbf[i] = f2bf(feat[i]);
  if (i < M_SIZE) {
    float x = pos[i * 3 + 0], y = pos[i * 3 + 1], z = pos[i * 3 + 2];
    float inv = 1.0f / sqrtf(x * x + y * y + z * z);
    posn[i * 3 + 0] = x * inv; posn[i * 3 + 1] = y * inv; posn[i * 3 + 2] = z * inv;
    float a = pemb[i * 3 + 0], b = pemb[i * 3 + 1], c = pemb[i * 3 + 2];
    float inv2 = 1.0f / sqrtf(a * a + b * b + c * c);
    embn[i * 3 + 0] = a * inv2; embn[i * 3 + 1] = b * inv2; embn[i * 3 + 2] = c * inv2;
  }
}

// Shared inner WMMA burst: loads 4 distinct B tiles (128B of one row) into
// separate register blocks so global-load latency overlaps 8 WMMAs, then
// prefetches the same row segment of the next K-chunk into the WGP cache
// (locality 3 -> WGP-scope global_prefetch_b8: pull into all cache levels).
#define WMMA_T_ITER(SRC, jc, t)                                              \
  {                                                                          \
    const u16* bcol = (SRC) + (size_t)((jc) + (t) * 32) * F_DIM + fb + (lane & 15); \
    V16 B0, B1, B2, B3;                                                      \
    for (int e = 0; e < 16; ++e) {                                           \
      const size_t kk = (size_t)((lane >> 4) * 8 + (e & 7) + (e >> 3) * 16) * F_DIM; \
      B0.h[e] = bcol[kk + 0];  B1.h[e] = bcol[kk + 16];                      \
      B2.h[e] = bcol[kk + 32]; B3.h[e] = bcol[kk + 48];                      \
    }                                                                        \
    __builtin_prefetch(                                                      \
        (SRC) + (size_t)((((jc) + 256) & (M_SIZE - 1)) + (t) * 32 + lane) *  \
                    F_DIM + fb, 0, 3);                                       \
    V16 Af0, Af1;                                                            \
    Af0.q[0] = pb[0][(t) * 32 + lane][0];                                    \
    Af0.q[1] = pb[0][(t) * 32 + lane][1];                                    \
    Af1.q[0] = pb[1][(t) * 32 + lane][0];                                    \
    Af1.q[1] = pb[1][(t) * 32 + lane][1];                                    \
    acc0[0] = __builtin_amdgcn_wmma_f32_16x16x32_bf16(                       \
        false, Af0.v, false, B0.v, (short)0, acc0[0], false, false);         \
    acc1[0] = __builtin_amdgcn_wmma_f32_16x16x32_bf16(                       \
        false, Af1.v, false, B0.v, (short)0, acc1[0], false, false);         \
    acc0[1] = __builtin_amdgcn_wmma_f32_16x16x32_bf16(                       \
        false, Af0.v, false, B1.v, (short)0, acc0[1], false, false);         \
    acc1[1] = __builtin_amdgcn_wmma_f32_16x16x32_bf16(                       \
        false, Af1.v, false, B1.v, (short)0, acc1[1], false, false);         \
    acc0[2] = __builtin_amdgcn_wmma_f32_16x16x32_bf16(                       \
        false, Af0.v, false, B2.v, (short)0, acc0[2], false, false);         \
    acc1[2] = __builtin_amdgcn_wmma_f32_16x16x32_bf16(                       \
        false, Af1.v, false, B2.v, (short)0, acc1[2], false, false);         \
    acc0[3] = __builtin_amdgcn_wmma_f32_16x16x32_bf16(                       \
        false, Af0.v, false, B3.v, (short)0, acc0[3], false, false);         \
    acc1[3] = __builtin_amdgcn_wmma_f32_16x16x32_bf16(                       \
        false, Af1.v, false, B3.v, (short)0, acc1[3], false, false);         \
  }

// ---------------------------------------------------------------------------
// Kernel 2: memory = softmax(posn @ embn^T) @ features  (bf16 out)
//           newpos = softmax(...) @ positions           (f32 out)
// Fused flash-style: logits are cosine sims in [-1,1], so exp() never
// overflows and no online max/rescale is needed.
// One block = 32 rows of M. 8 waves; wave w owns feature columns [64w,64w+64).
// Each wave builds the A-fragment (ISA 16-bit 16x32 layout) for its own
// 32-column K sub-chunk and shares it through LDS as raw dwords.
// ---------------------------------------------------------------------------
__global__ __launch_bounds__(256)
void build_memory_kernel(const u16*  __restrict__ featbf,
                         const float* __restrict__ posn,
                         const float* __restrict__ embn,
                         const float* __restrict__ pos_raw,
                         u16*  __restrict__ membf,
                         float* __restrict__ newpos) {
  __shared__ uint4 pb[2][256][2];        // [m-frag][wave*32+lane][half] = A frags
  __shared__ float s_lp[8][32];          // per-wave denominator partials
  __shared__ float s_npp[8][32][3];      // per-wave attn@positions partials
  __shared__ float s_l[32];              // final denominators

  const int tid   = threadIdx.x;
  const int wv    = tid >> 5;
  const int lane  = tid & 31;
  const int half  = lane >> 4;           // 0: K 0..7/16..23   1: K 8..15/24..31
  const int ml    = lane & 15;           // A-fragment row within 16-tile
  const int mbase = blockIdx.x * 32;
  const int fb    = wv * 64;

  // per-lane K offsets of the 16 A-fragment elements (ISA 16-bit A layout)
  int koff[16];
#pragma unroll
  for (int e = 0; e < 16; ++e) koff[e] = half * 8 + (e & 7) + (e >> 3) * 16;

  const float a0x = posn[(mbase + ml) * 3 + 0];
  const float a0y = posn[(mbase + ml) * 3 + 1];
  const float a0z = posn[(mbase + ml) * 3 + 2];
  const float a1x = posn[(mbase + 16 + ml) * 3 + 0];
  const float a1y = posn[(mbase + 16 + ml) * 3 + 1];
  const float a1z = posn[(mbase + 16 + ml) * 3 + 2];

  v8f acc0[4], acc1[4];
#pragma unroll
  for (int u = 0; u < 4; ++u) {
    acc0[u] = (v8f){0.f, 0.f, 0.f, 0.f, 0.f, 0.f, 0.f, 0.f};
    acc1[u] = (v8f){0.f, 0.f, 0.f, 0.f, 0.f, 0.f, 0.f, 0.f};
  }
  float l0 = 0.f, l1 = 0.f;
  float np0[3] = {0.f, 0.f, 0.f}, np1[3] = {0.f, 0.f, 0.f};

  for (int jc = 0; jc < M_SIZE; jc += 256) {
    const int jb = jc + wv * 32;
    V16 A0, A1;
#pragma unroll
    for (int e = 0; e < 16; ++e) {
      const int j = jb + koff[e];
      const float ex = embn[j * 3 + 0], ey = embn[j * 3 + 1], ez = embn[j * 3 + 2];
      const float s0 = a0x * ex + a0y * ey + a0z * ez;   // in [-1,1]
      const float s1 = a1x * ex + a1y * ey + a1z * ez;
      const float p0 = __expf(s0);
      const float p1 = __expf(s1);
      A0.h[e] = f2bf(p0);
      A1.h[e] = f2bf(p1);
      l0 += p0; l1 += p1;
      const float px = pos_raw[j * 3 + 0], py = pos_raw[j * 3 + 1], pz = pos_raw[j * 3 + 2];
      np0[0] += p0 * px; np0[1] += p0 * py; np0[2] += p0 * pz;
      np1[0] += p1 * px; np1[1] += p1 * py; np1[2] += p1 * pz;
    }
    pb[0][wv * 32 + lane][0] = A0.q[0];
    pb[0][wv * 32 + lane][1] = A0.q[1];
    pb[1][wv * 32 + lane][0] = A1.q[0];
    pb[1][wv * 32 + lane][1] = A1.q[1];
    __syncthreads();

#pragma unroll
    for (int t = 0; t < 8; ++t) {
      WMMA_T_ITER(featbf, jc, t)
    }
    __syncthreads();
  }

  // Deterministic reduction of denominators and attn@positions.
  l0 += __shfl_xor(l0, 16);
  l1 += __shfl_xor(l1, 16);
#pragma unroll
  for (int c = 0; c < 3; ++c) {
    np0[c] += __shfl_xor(np0[c], 16);
    np1[c] += __shfl_xor(np1[c], 16);
  }
  if (half == 0) {
    s_lp[wv][ml]      = l0;
    s_lp[wv][16 + ml] = l1;
#pragma unroll
    for (int c = 0; c < 3; ++c) {
      s_npp[wv][ml][c]      = np0[c];
      s_npp[wv][16 + ml][c] = np1[c];
    }
  }
  __syncthreads();
  if (tid < 32) {
    float L = 0.f, N0 = 0.f, N1 = 0.f, N2 = 0.f;
#pragma unroll
    for (int w = 0; w < 8; ++w) {
      L  += s_lp[w][tid];
      N0 += s_npp[w][tid][0];
      N1 += s_npp[w][tid][1];
      N2 += s_npp[w][tid][2];
    }
    s_l[tid] = L;
    const float inv = 1.0f / L;
    newpos[(mbase + tid) * 3 + 0] = N0 * inv;
    newpos[(mbase + tid) * 3 + 1] = N1 * inv;
    newpos[(mbase + tid) * 3 + 2] = N2 * inv;
  }
  __syncthreads();

  // C/D layout: VGPR r, lanes 0-15 -> M=r, N=lane; lanes 16-31 -> M=8+r.
#pragma unroll
  for (int r = 0; r < 8; ++r) {
    const int m0   = r + half * 8;
    const float i0 = 1.0f / s_l[m0];
    const float i1 = 1.0f / s_l[16 + m0];
    const size_t row0 = (size_t)(mbase + m0) * F_DIM + fb + ml;
    const size_t row1 = (size_t)(mbase + 16 + m0) * F_DIM + fb + ml;
#pragma unroll
    for (int u = 0; u < 4; ++u) {
      membf[row0 + u * 16] = f2bf(acc0[u][r] * i0);
      membf[row1 + u * 16] = f2bf(acc1[u][r] * i1);
    }
  }
}

// ---------------------------------------------------------------------------
// Kernel 3: retrieved = softmax(-cdist(q, newpos)) @ memory   (f32 out)
// Logits -dist <= 0, so exp() never overflows; same fused structure.
// One block = 32 queries; wave w owns feature columns [64w, 64w+64).
// ---------------------------------------------------------------------------
__global__ __launch_bounds__(256)
void query_kernel(const u16*  __restrict__ membf,
                  const float* __restrict__ newpos,
                  const float* __restrict__ qpos,
                  float* __restrict__ out) {
  __shared__ uint4 pb[2][256][2];
  __shared__ float s_lp[8][32];
  __shared__ float s_l[32];

  const int tid   = threadIdx.x;
  const int wv    = tid >> 5;
  const int lane  = tid & 31;
  const int half  = lane >> 4;
  const int ml    = lane & 15;
  const int qbase = blockIdx.x * 32;
  const int fb    = wv * 64;

  int koff[16];
#pragma unroll
  for (int e = 0; e < 16; ++e) koff[e] = half * 8 + (e & 7) + (e >> 3) * 16;

  const float q0x = qpos[(qbase + ml) * 3 + 0];
  const float q0y = qpos[(qbase + ml) * 3 + 1];
  const float q0z = qpos[(qbase + ml) * 3 + 2];
  const float q1x = qpos[(qbase + 16 + ml) * 3 + 0];
  const float q1y = qpos[(qbase + 16 + ml) * 3 + 1];
  const float q1z = qpos[(qbase + 16 + ml) * 3 + 2];
  const float q0s = q0x * q0x + q0y * q0y + q0z * q0z;
  const float q1s = q1x * q1x + q1y * q1y + q1z * q1z;

  v8f acc0[4], acc1[4];
#pragma unroll
  for (int u = 0; u < 4; ++u) {
    acc0[u] = (v8f){0.f, 0.f, 0.f, 0.f, 0.f, 0.f, 0.f, 0.f};
    acc1[u] = (v8f){0.f, 0.f, 0.f, 0.f, 0.f, 0.f, 0.f, 0.f};
  }
  float l0 = 0.f, l1 = 0.f;

  for (int jc = 0; jc < M_SIZE; jc += 256) {
    const int jb = jc + wv * 32;
    V16 A0, A1;
#pragma unroll
    for (int e = 0; e < 16; ++e) {
      const int j = jb + koff[e];
      const float nx = newpos[j * 3 + 0], ny = newpos[j * 3 + 1], nz = newpos[j * 3 + 2];
      const float n2 = nx * nx + ny * ny + nz * nz;
      const float d20 = q0s + n2 - 2.0f * (q0x * nx + q0y * ny + q0z * nz);
      const float d21 = q1s + n2 - 2.0f * (q1x * nx + q1y * ny + q1z * nz);
      const float p0 = __expf(-sqrtf(fmaxf(d20, 1e-12f)));
      const float p1 = __expf(-sqrtf(fmaxf(d21, 1e-12f)));
      A0.h[e] = f2bf(p0);
      A1.h[e] = f2bf(p1);
      l0 += p0; l1 += p1;
    }
    pb[0][wv * 32 + lane][0] = A0.q[0];
    pb[0][wv * 32 + lane][1] = A0.q[1];
    pb[1][wv * 32 + lane][0] = A1.q[0];
    pb[1][wv * 32 + lane][1] = A1.q[1];
    __syncthreads();

#pragma unroll
    for (int t = 0; t < 8; ++t) {
      WMMA_T_ITER(membf, jc, t)
    }
    __syncthreads();
  }

  l0 += __shfl_xor(l0, 16);
  l1 += __shfl_xor(l1, 16);
  if (half == 0) {
    s_lp[wv][ml]      = l0;
    s_lp[wv][16 + ml] = l1;
  }
  __syncthreads();
  if (tid < 32) {
    float L = 0.f;
#pragma unroll
    for (int w = 0; w < 8; ++w) L += s_lp[w][tid];
    s_l[tid] = L;
  }
  __syncthreads();

#pragma unroll
  for (int r = 0; r < 8; ++r) {
    const int m0   = r + half * 8;
    const float i0 = 1.0f / s_l[m0];
    const float i1 = 1.0f / s_l[16 + m0];
    const size_t row0 = (size_t)(qbase + m0) * F_DIM + fb + ml;
    const size_t row1 = (size_t)(qbase + 16 + m0) * F_DIM + fb + ml;
#pragma unroll
    for (int u = 0; u < 4; ++u) {
      out[row0 + u * 16] = acc0[u][r] * i0;
      out[row1 + u * 16] = acc1[u][r] * i1;
    }
  }
}

// ---------------------------------------------------------------------------
// Launch. Workspace layout (bytes):
//   [0)        posn    8192*3*4   = 98304
//   [98304)    embn    8192*3*4   = 98304
//   [196608)   newpos  8192*3*4   = 98304
//   [294912)   featbf  8192*512*2 = 8388608
//   [8683520)  membf   8192*512*2 = 8388608
//   total ~17.1 MB
// ---------------------------------------------------------------------------
extern "C" void kernel_launch(void* const* d_in, const int* in_sizes, int n_in,
                              void* d_out, int out_size, void* d_ws, size_t ws_size,
                              hipStream_t stream) {
  const float* features  = (const float*)d_in[0];  // [8192, 512]
  const float* positions = (const float*)d_in[1];  // [8192, 3]
  const float* qpos      = (const float*)d_in[2];  // [32768, 3]
  const float* pemb      = (const float*)d_in[3];  // [8192, 3]
  float*       out       = (float*)d_out;          // [32768, 512]

  char* ws = (char*)d_ws;
  float* posn   = (float*)(ws + 0);
  float* embn   = (float*)(ws + 98304);
  float* newpos = (float*)(ws + 196608);
  u16*   featbf = (u16*)(ws + 294912);
  u16*   membf  = (u16*)(ws + 8683520);

  prep_kernel<<<(M_SIZE * F_DIM + 255) / 256, 256, 0, stream>>>(
      features, positions, pemb, featbf, posn, embn);

  build_memory_kernel<<<M_SIZE / 32, 256, 0, stream>>>(
      featbf, posn, embn, positions, membf, newpos);

  query_kernel<<<N_Q / 32, 256, 0, stream>>>(
      membf, newpos, qpos, out);
}
